// EncoderLayer_85100482003492
// MI455X (gfx1250) — matrix operand order repaired and checked by reading, not compiled, not run on hardware
//
#include <hip/hip_runtime.h>
#include <hip/hip_bf16.h>
#include <stdint.h>

// Problem constants (from reference)
#define B_    4
#define S_    4096
#define D_    512
#define H_    8
#define DFF_  2048
#define DEG_  32
#define DH_   64
#define M_    (B_ * S_)     // 16384 rows for all GEMMs
#define EPS_  1e-6f

typedef __attribute__((ext_vector_type(16))) __bf16 bf16x16;
typedef __attribute__((ext_vector_type(8)))  float  fx8;

union FragU {
    bf16x16 v;
    unsigned short u[16];
    uint4 q[2];
};

__device__ __forceinline__ unsigned short f2bf(float f) {
    union { float f; unsigned u; } c; c.f = f;
    unsigned u = c.u;
    u += 0x7FFFu + ((u >> 16) & 1u);   // round-to-nearest-even
    return (unsigned short)(u >> 16);
}

// ---------------------------------------------------------------------------
// elementwise f32 -> bf16 cast
// ---------------------------------------------------------------------------
__global__ void cast_f32_bf16_kernel(const float* __restrict__ src,
                                     unsigned short* __restrict__ dst, int n) {
    int i = blockIdx.x * blockDim.x + threadIdx.x;
    int stride = gridDim.x * blockDim.x;
    for (; i < n; i += stride) dst[i] = f2bf(src[i]);
}

// ---------------------------------------------------------------------------
// bf16 WMMA GEMM:  C[M,N] = A[M,K] @ B[K,N] + bias[N]
// Block tile 128x128, BK=32, 8 waves; wave = 64x32 (4x2 16x16 frags).
// A tile staged with GLOBAL_LOAD_ASYNC_TO_LDS_B128 (ASYNCcnt-tracked, no VGPR
// round-trip); B tile staged through VGPRs because it is transposed into LDS.
// RELU_BF16==0: store f32 to Cf.  RELU_BF16==1: relu then bf16 store to Cb.
// ---------------------------------------------------------------------------
template <int RELU_BF16>
__global__ __launch_bounds__(256)
void gemm_bf16_kernel(const unsigned short* __restrict__ A,
                      const unsigned short* __restrict__ Bw,
                      const float* __restrict__ bias,
                      float* __restrict__ Cf,
                      unsigned short* __restrict__ Cb,
                      int Mdim, int Ndim, int Kdim) {
    __shared__ __align__(16) unsigned short As[128 * 32];   // [row][k]
    __shared__ __align__(16) unsigned short Bs[128 * 32];   // transposed: [n][k]

    const int t    = threadIdx.x;
    const int lane = t & 31;
    const int wid  = t >> 5;
    const int wr   = wid >> 2;     // 0..1 : 64-row group
    const int wc   = wid & 3;      // 0..3 : 32-col group
    const int half = lane >> 4;    // lane group (K-half selector)
    const int l16  = lane & 15;
    const int m0   = blockIdx.y * 128;
    const int n0   = blockIdx.x * 128;

    fx8 acc[4][2] = {};

    const int ksteps = Kdim >> 5;
    for (int kk = 0; kk < ksteps; ++kk) {
        const int k0 = kk << 5;
        // --- stage A tile (128x32 bf16): async global->LDS, 16B per lane ---
        {
            int f = t * 8;
            int r = f >> 5, c = f & 31;
#if defined(__gfx1250__)
            {
                unsigned lds0 = (unsigned)(uintptr_t)&As[r * 32 + c];
                unsigned g0   = (unsigned)(((size_t)(m0 + r) * Kdim + k0 + c) * 2);
                asm volatile("global_load_async_to_lds_b128 %0, %1, %2"
                             :: "v"(lds0), "v"(g0),
                                "s"((unsigned long long)(uintptr_t)A)
                             : "memory");
                f += 2048; r = f >> 5; c = f & 31;
                unsigned lds1 = (unsigned)(uintptr_t)&As[r * 32 + c];
                unsigned g1   = (unsigned)(((size_t)(m0 + r) * Kdim + k0 + c) * 2);
                asm volatile("global_load_async_to_lds_b128 %0, %1, %2"
                             :: "v"(lds1), "v"(g1),
                                "s"((unsigned long long)(uintptr_t)A)
                             : "memory");
            }
#else
            *(uint4*)&As[r * 32 + c] =
                *(const uint4*)&A[(size_t)(m0 + r) * Kdim + k0 + c];
            f += 2048; r = f >> 5; c = f & 31;
            *(uint4*)&As[r * 32 + c] =
                *(const uint4*)&A[(size_t)(m0 + r) * Kdim + k0 + c];
#endif
        }
        // --- stage B tile transposed to [n][k] for vector frag reads ---
        {
            int f = t * 8;
            int kr = f >> 7, nc = f & 127;
            union { uint4 q; unsigned short s[8]; } ld;
            ld.q = *(const uint4*)&Bw[(size_t)(k0 + kr) * Ndim + n0 + nc];
            #pragma unroll
            for (int u = 0; u < 8; ++u) Bs[(nc + u) * 32 + kr] = ld.s[u];
            kr += 16;
            ld.q = *(const uint4*)&Bw[(size_t)(k0 + kr) * Ndim + n0 + nc];
            #pragma unroll
            for (int u = 0; u < 8; ++u) Bs[(nc + u) * 32 + kr] = ld.s[u];
        }
#if defined(__gfx1250__)
        asm volatile("s_wait_asynccnt 0x0" ::: "memory");
#endif
        __syncthreads();

        // B frags: lane l16 = column n; elems 0..7 <-> k = half*8 + j,
        //          elems 8..15 <-> k = 16 + half*8 + (j-8)
        FragU fb[2];
        #pragma unroll
        for (int j = 0; j < 2; ++j) {
            const int nb = wc * 32 + j * 16 + l16;
            fb[j].q[0] = *(const uint4*)&Bs[nb * 32 + half * 8];
            fb[j].q[1] = *(const uint4*)&Bs[nb * 32 + 16 + half * 8];
        }
        // A frags: lane l16 = row m; elems 0..7 <-> k = half*8 + j,
        //          elems 8..15 <-> k = 16 + half*8 + (j-8)
        #pragma unroll
        for (int i = 0; i < 4; ++i) {
            FragU fa;
            const int mb = wr * 64 + i * 16 + l16;
            fa.q[0] = *(const uint4*)&As[mb * 32 + half * 8];
            fa.q[1] = *(const uint4*)&As[mb * 32 + 16 + half * 8];
            #pragma unroll
            for (int j = 0; j < 2; ++j) {
                acc[i][j] = __builtin_amdgcn_wmma_f32_16x16x32_bf16(
                    false, fa.v, false, fb[j].v, (short)0, acc[i][j],
                    false, false);
            }
        }
        __syncthreads();
    }

    // epilogue: C/D layout -> lane l16 = col, elem r -> row half*8 + r
    #pragma unroll
    for (int i = 0; i < 4; ++i) {
        #pragma unroll
        for (int j = 0; j < 2; ++j) {
            const int col = n0 + wc * 32 + j * 16 + l16;
            const float bv = bias[col];
            #pragma unroll
            for (int r = 0; r < 8; ++r) {
                const int row = m0 + wr * 64 + i * 16 + half * 8 + r;
                float val = acc[i][j][r] + bv;
                if (RELU_BF16) {
                    val = val > 0.f ? val : 0.f;
                    Cb[(size_t)row * Ndim + col] = f2bf(val);
                } else {
                    Cf[(size_t)row * Ndim + col] = val;
                }
            }
        }
    }
}

// ---------------------------------------------------------------------------
// Edge attention: block per (b,s); wave per head; lane per edge (DEG==32).
// Contiguous segments: edges [s*32,(s+1)*32) belong to query s.
// ---------------------------------------------------------------------------
__global__ __launch_bounds__(256)
void attn_kernel(const float* __restrict__ q, const float* __restrict__ k,
                 const float* __restrict__ v, const int* __restrict__ edges,
                 unsigned short* __restrict__ ctxb) {
    __shared__ __align__(16) float qs[D_];
    const int blk = blockIdx.x;
    const int b = blk / S_;
    const int s = blk - b * S_;
    const int t = threadIdx.x;
    const int h = t >> 5;         // wave == head
    const int ln = t & 31;        // lane == edge index (and output d index)

    const size_t rowq = ((size_t)b * S_ + s) * D_;
    qs[t] = q[rowq + t];
    qs[t + 256] = q[rowq + t + 256];
    __syncthreads();

    const int ki = edges[((size_t)s * DEG_ + ln) * 2 + 1];
    const float* kp = k + ((size_t)b * S_ + ki) * D_ + h * DH_;
    const float* qp = qs + h * DH_;
    float dot = 0.f;
    #pragma unroll
    for (int d = 0; d < DH_; d += 4) {
        float4 kq = *(const float4*)&kp[d];
        float4 qq = *(const float4*)&qp[d];
        dot += qq.x * kq.x + qq.y * kq.y + qq.z * kq.z + qq.w * kq.w;
    }
    const float sc = dot * 0.125f;  // 1/sqrt(DH)

    // wave32 reductions: max then sum (one head per wave)
    float m = sc;
    #pragma unroll
    for (int off = 16; off >= 1; off >>= 1) {
        float o = __shfl_xor(m, off);
        m = m > o ? m : o;
    }
    float p = __expf(sc - m);
    float den = p;
    #pragma unroll
    for (int off = 16; off >= 1; off >>= 1) den += __shfl_xor(den, off);
    p = p / den;

    // ctx[h*64 + d] = sum_e p[e] * v[b, ki[e], h*64 + d]; lane ln does d=ln, ln+32
    float c0 = 0.f, c1 = 0.f;
    for (int e = 0; e < 32; ++e) {
        const float pe = __shfl(p, e);
        const int kie  = __shfl(ki, e);
        const float* vp = v + ((size_t)b * S_ + kie) * D_ + h * DH_;
        c0 += pe * vp[ln];
        c1 += pe * vp[ln + 32];
    }
    const size_t orow = ((size_t)b * S_ + s) * D_ + h * DH_;
    ctxb[orow + ln]      = f2bf(c0);
    ctxb[orow + ln + 32] = f2bf(c1);
}

// ---------------------------------------------------------------------------
// y = LayerNorm(a + c) * g + beta ; optional f32 and bf16 outputs.
// One block per row, 256 threads, D=512 (2 elems/thread).
// ---------------------------------------------------------------------------
__global__ __launch_bounds__(256)
void ln_add_kernel(const float* __restrict__ a, const float* __restrict__ c,
                   const float* __restrict__ g, const float* __restrict__ beta,
                   float* __restrict__ yf, unsigned short* __restrict__ yb) {
    __shared__ float rs[256];
    __shared__ float rs2[256];
    const int row = blockIdx.x;
    const int t = threadIdx.x;
    const size_t base = (size_t)row * D_;
    const float v0 = a[base + t] + c[base + t];
    const float v1 = a[base + t + 256] + c[base + t + 256];
    rs[t]  = v0 + v1;
    rs2[t] = v0 * v0 + v1 * v1;
    __syncthreads();
    for (int off = 128; off >= 1; off >>= 1) {
        if (t < off) { rs[t] += rs[t + off]; rs2[t] += rs2[t + off]; }
        __syncthreads();
    }
    const float mu  = rs[0] * (1.0f / D_);
    const float var = rs2[0] * (1.0f / D_) - mu * mu;
    const float inv = rsqrtf(var + EPS_);
    const float y0 = (v0 - mu) * inv * g[t] + beta[t];
    const float y1 = (v1 - mu) * inv * g[t + 256] + beta[t + 256];
    if (yf) { yf[base + t] = y0; yf[base + t + 256] = y1; }
    if (yb) { yb[base + t] = f2bf(y0); yb[base + t + 256] = f2bf(y1); }
}

// ---------------------------------------------------------------------------
extern "C" void kernel_launch(void* const* d_in, const int* in_sizes, int n_in,
                              void* d_out, int out_size, void* d_ws, size_t ws_size,
                              hipStream_t stream) {
    (void)in_sizes; (void)n_in; (void)out_size; (void)ws_size;
    const float* x     = (const float*)d_in[0];
    const int*   edges = (const int*)  d_in[1];
    const float* wq    = (const float*)d_in[2];
    const float* bq    = (const float*)d_in[3];
    const float* wk    = (const float*)d_in[4];
    const float* bk    = (const float*)d_in[5];
    const float* wv    = (const float*)d_in[6];
    const float* bv    = (const float*)d_in[7];
    const float* wo    = (const float*)d_in[8];
    const float* bo    = (const float*)d_in[9];
    const float* w1    = (const float*)d_in[10];
    const float* b1    = (const float*)d_in[11];
    const float* w2    = (const float*)d_in[12];
    const float* b2    = (const float*)d_in[13];
    const float* ln1g  = (const float*)d_in[14];
    const float* ln1b  = (const float*)d_in[15];
    const float* ln2g  = (const float*)d_in[16];
    const float* ln2b  = (const float*)d_in[17];

    char* ws = (char*)d_ws;
    size_t off = 0;
    auto alloc = [&](size_t bytes) -> void* {
        void* p = ws + off;
        off += (bytes + 255) & ~(size_t)255;
        return p;
    };
    unsigned short* xb   = (unsigned short*)alloc((size_t)M_ * D_ * 2);
    unsigned short* wqb  = (unsigned short*)alloc((size_t)D_ * D_ * 2);
    unsigned short* wkb  = (unsigned short*)alloc((size_t)D_ * D_ * 2);
    unsigned short* wvb  = (unsigned short*)alloc((size_t)D_ * D_ * 2);
    unsigned short* wob  = (unsigned short*)alloc((size_t)D_ * D_ * 2);
    unsigned short* w1b  = (unsigned short*)alloc((size_t)D_ * DFF_ * 2);
    unsigned short* w2b  = (unsigned short*)alloc((size_t)DFF_ * D_ * 2);
    float*          qf   = (float*)alloc((size_t)M_ * D_ * 4);
    float*          kf   = (float*)alloc((size_t)M_ * D_ * 4);
    float*          vf   = (float*)alloc((size_t)M_ * D_ * 4);
    unsigned short* ctxb = (unsigned short*)alloc((size_t)M_ * D_ * 2);
    float*          attnf= (float*)alloc((size_t)M_ * D_ * 4);
    unsigned short* hb   = (unsigned short*)alloc((size_t)M_ * DFF_ * 2);
    // aliases (producers strictly after last consumer of the aliased buffer)
    float*          x1f  = qf;      // LN1 f32 out reuses q
    unsigned short* x1b  = ctxb;    // LN1 bf16 out reuses ctx
    float*          ff   = attnf;   // FFN GEMM2 out reuses attn

    const int ct = 256;
    auto cgrid = [](size_t n) { return (unsigned)((n + 255) / 256); };

    // casts
    cast_f32_bf16_kernel<<<cgrid((size_t)M_ * D_), ct, 0, stream>>>(x, xb, M_ * D_);
    cast_f32_bf16_kernel<<<cgrid((size_t)D_ * D_), ct, 0, stream>>>(wq, wqb, D_ * D_);
    cast_f32_bf16_kernel<<<cgrid((size_t)D_ * D_), ct, 0, stream>>>(wk, wkb, D_ * D_);
    cast_f32_bf16_kernel<<<cgrid((size_t)D_ * D_), ct, 0, stream>>>(wv, wvb, D_ * D_);
    cast_f32_bf16_kernel<<<cgrid((size_t)D_ * D_), ct, 0, stream>>>(wo, wob, D_ * D_);
    cast_f32_bf16_kernel<<<cgrid((size_t)D_ * DFF_), ct, 0, stream>>>(w1, w1b, D_ * DFF_);
    cast_f32_bf16_kernel<<<cgrid((size_t)DFF_ * D_), ct, 0, stream>>>(w2, w2b, DFF_ * D_);

    // QKV projections
    dim3 gD(D_ / 128, M_ / 128);
    gemm_bf16_kernel<0><<<gD, 256, 0, stream>>>(xb, wqb, bq, qf, nullptr, M_, D_, D_);
    gemm_bf16_kernel<0><<<gD, 256, 0, stream>>>(xb, wkb, bk, kf, nullptr, M_, D_, D_);
    gemm_bf16_kernel<0><<<gD, 256, 0, stream>>>(xb, wvb, bv, vf, nullptr, M_, D_, D_);

    // edge attention -> ctx (bf16)
    attn_kernel<<<M_, 256, 0, stream>>>(qf, kf, vf, edges, ctxb);

    // output projection (+bo)
    gemm_bf16_kernel<0><<<gD, 256, 0, stream>>>(ctxb, wob, bo, attnf, nullptr, M_, D_, D_);

    // LN1(x + attn) -> x1 (f32 + bf16)
    ln_add_kernel<<<M_, 256, 0, stream>>>(x, attnf, ln1g, ln1b, x1f, x1b);

    // FFN
    dim3 gF(DFF_ / 128, M_ / 128);
    gemm_bf16_kernel<1><<<gF, 256, 0, stream>>>(x1b, w1b, b1, nullptr, hb, M_, DFF_, D_);
    gemm_bf16_kernel<0><<<gD, 256, 0, stream>>>(hb, w2b, b2, ff, nullptr, M_, D_, DFF_);

    // LN2(x1 + ffn) -> out
    ln_add_kernel<<<M_, 256, 0, stream>>>(x1f, ff, ln2g, ln2b, (float*)d_out, nullptr);
}
